// Structure_aware_THTN3_16552803959370
// MI455X (gfx1250) — matrix operand order repaired and checked by reading, not compiled, not run on hardware
//
#include <hip/hip_runtime.h>
#include <hip/hip_bf16.h>
#include <math.h>

// ---------------------------------------------------------------------------
// Types for CDNA5 WMMA
// ---------------------------------------------------------------------------
typedef _Float16 v16h __attribute__((ext_vector_type(16)));
typedef float    v8f  __attribute__((ext_vector_type(8)));

#define DDIM      128     // feature dim everywhere (D == Q == INP == hidden)
#define TILE_ROWS 128     // rows per block in the WMMA linear kernel
#define NTHREADS  256     // 8 wave32 per block

// ---------------------------------------------------------------------------
// Fragment-position helper (CDNA5 WMMA f16 VGPR layouts, ISA 7.12.2)
// A (16x32, MxK): lanes 0-15 hold M=lane, K groups {0..7,16..23};
//                 lanes 16-31 hold M=lane-16, K groups {8..15,24..31}.
// ---------------------------------------------------------------------------
__device__ __forceinline__ int a_frag_pos(int m, int kk) {
  int hi   = (kk >> 3) & 1;          // which lane half holds this K
  int base = kk - 8 * hi;            // 0..7 or 16..23
  int j    = (base < 8) ? (base >> 1) : (4 + ((base - 16) >> 1)); // VGPR index
  int p    = base & 1;               // half within dword
  int lane = m + 16 * hi;
  return (lane << 4) + 2 * j + p;    // index into [32 lanes][16 halves]
}

// ---------------------------------------------------------------------------
// WMMA linear: out[M, O] = X[M,128] @ W[O,128]^T + bias (+ epilogue)
// MODE 0: bias only
// MODE 1: bias + add1[row] + tab1[idx1[row]] + tab2[idx2[row]]  (node features)
// MODE 2: bias then ReLU                                         (FFN hidden)
// NCT: number of 16-col output tiles (compile-time, fully unrolled)
// ---------------------------------------------------------------------------
template<int MODE, int NCT>
__global__ __launch_bounds__(NTHREADS)
void wmma_linear(const float* __restrict__ X, int M,
                 const float* __restrict__ W, int Oreal,
                 const float* __restrict__ bias,
                 float* __restrict__ out, int ostride,
                 const float* __restrict__ add1,
                 const float* __restrict__ tab1, const int* __restrict__ idx1,
                 const float* __restrict__ tab2, const int* __restrict__ idx2)
{
  // [tile][kchunk(4)][lane(32)][half(16)] f16
  __shared__ __attribute__((aligned(32))) _Float16 lds_x[8 * 4 * 32 * 16];
  __shared__ __attribute__((aligned(32))) _Float16 lds_w[NCT * 4 * 32 * 16];

  const int tid     = threadIdx.x;
  const int rowbase = blockIdx.x * TILE_ROWS;

  // ---- stage X tile: f32 -> f16, written in A-fragment order ----
  for (int i = tid; i < TILE_ROWS * 32; i += NTHREADS) {
    int r    = i >> 5;                 // 0..127 row in tile
    int k4   = (i & 31) << 2;          // k base (float4 granularity)
    int grow = rowbase + r;
    float vv[4] = {0.f, 0.f, 0.f, 0.f};
    if (grow < M) {
      const float4 t = ((const float4*)(X + (size_t)grow * DDIM))[i & 31];
      vv[0] = t.x; vv[1] = t.y; vv[2] = t.z; vv[3] = t.w;
    }
    int rt = r >> 4, m = r & 15;
    #pragma unroll
    for (int u = 0; u < 4; ++u) {
      int k = k4 + u;
      int c = k >> 5, kk = k & 31;
      lds_x[((rt * 4 + c) << 9) + a_frag_pos(m, kk)] = (_Float16)vv[u];
    }
  }

  // ---- stage W: f32 -> f16 in B-fragment order; zero-pad cols >= Oreal ----
  for (int i = tid; i < NCT * 16 * DDIM; i += NTHREADS) {
    int o = i >> 7;                    // output column
    int k = i & 127;                   // reduction index
    float w = (NCT == 8 || o < Oreal) ? W[(size_t)o * DDIM + k] : 0.f;
    int ct = o >> 4, n = o & 15;
    int c  = k >> 5, kk = k & 31;
    int hi = (kk >> 3) & 1, j = ((kk >> 4) << 2) + ((kk & 7) >> 1), p = kk & 1;
    lds_w[((ct * 4 + c) << 9) + ((n + 16 * hi) << 4) + 2 * j + p] = (_Float16)w;
  }
  __syncthreads();

  // ---- compute: wave `rt` owns row-tile rt, loops over column tiles ----
  const int lane = tid & 31;
  const int rt   = tid >> 5;
  const int n    = lane & 15;
  const int mb   = (lane >> 4) << 3;          // 0 or 8 (C-matrix row offset)
  const bool full_rows = (rowbase + TILE_ROWS) <= M;   // block-uniform

  // preload all A fragments for this row tile (held in VGPRs across tiles)
  v16h a[4];
  #pragma unroll
  for (int c = 0; c < 4; ++c)
    a[c] = *(const v16h*)&lds_x[((rt * 4 + c) << 9) + (lane << 4)];

  #pragma unroll
  for (int ct = 0; ct < NCT; ++ct) {
    // preload all 4 B fragments, then run the WMMA chain back-to-back
    v16h b[4];
    #pragma unroll
    for (int c = 0; c < 4; ++c)
      b[c] = *(const v16h*)&lds_w[((ct * 4 + c) << 9) + (lane << 4)];

    v8f acc = {};
    #pragma unroll
    for (int c = 0; c < 4; ++c)        // K = 128 = 4 x 32
      acc = __builtin_amdgcn_wmma_f32_16x16x32_f16(
          /*neg_a=*/false, a[c], /*neg_b=*/false, b[c],
          /*c_mod=*/(short)0, acc, /*reuse_a=*/false, /*reuse_b=*/false);

    const int o = ct * 16 + n;
    if (NCT == 8 || o < Oreal) {
      const float bv = bias[o];
      if (full_rows) {
        #pragma unroll
        for (int r = 0; r < 8; ++r) {
          int grow = rowbase + rt * 16 + mb + r;
          float val = acc[r] + bv;
          if (MODE == 1) {
            val += add1[(size_t)grow * DDIM + o];
            val += tab1[(size_t)idx1[grow] * DDIM + o];
            val += tab2[(size_t)idx2[grow] * DDIM + o];
          }
          if (MODE == 2) val = fmaxf(val, 0.f);
          out[(size_t)grow * ostride + o] = val;
        }
      } else {
        #pragma unroll
        for (int r = 0; r < 8; ++r) {
          int grow = rowbase + rt * 16 + mb + r;
          if (grow < M) {
            float val = acc[r] + bv;
            if (MODE == 1) {
              val += add1[(size_t)grow * DDIM + o];
              val += tab1[(size_t)idx1[grow] * DDIM + o];
              val += tab2[(size_t)idx2[grow] * DDIM + o];
            }
            if (MODE == 2) val = fmaxf(val, 0.f);
            out[(size_t)grow * ostride + o] = val;
          }
        }
      }
    }
  }
}

// ---------------------------------------------------------------------------
// Fill kernel (init atomic targets)
// ---------------------------------------------------------------------------
__global__ void fill_kernel(float* __restrict__ p, float v, size_t n) {
  size_t i = (size_t)blockIdx.x * blockDim.x + threadIdx.x;
  size_t stride = (size_t)gridDim.x * blockDim.x;
  for (; i < n; i += stride) p[i] = v;
}

// float atomic max via the classic signed-max / unsigned-min trick
__device__ __forceinline__ void atomicMaxFloat(float* addr, float v) {
  if (v >= 0.f) atomicMax((int*)addr, __float_as_int(v));
  else          atomicMin((unsigned int*)addr, __float_as_uint(v));
}

// ---------------------------------------------------------------------------
// score = leaky_relu(dot(A[ia[e]], B[ib[e]])) * scale + addv[e]
// also: segmax[idst[e]] = max(segmax, score)   (wave per entry)
// ---------------------------------------------------------------------------
__global__ __launch_bounds__(NTHREADS)
void score_max_kernel(const float* __restrict__ A, const float* __restrict__ B,
                      const int* __restrict__ ia, const int* __restrict__ ib,
                      const int* __restrict__ idst,
                      const float* __restrict__ addv,
                      float* __restrict__ score, float* __restrict__ segmax,
                      int nnz, float scale)
{
  int e    = blockIdx.x * (NTHREADS >> 5) + (threadIdx.x >> 5);
  int lane = threadIdx.x & 31;
  if (e >= nnz) return;
  int ra = ia[e], rb = ib[e];
  const float4 a = ((const float4*)(A + (size_t)ra * DDIM))[lane];
  const float4 b = ((const float4*)(B + (size_t)rb * DDIM))[lane];
  float part = a.x * b.x + a.y * b.y + a.z * b.z + a.w * b.w;
  #pragma unroll
  for (int off = 16; off > 0; off >>= 1) part += __shfl_down(part, off, 32);
  if (lane == 0) {
    float sc = (part > 0.f) ? part : 0.01f * part;   // leaky_relu slope 0.01
    sc = sc * scale + addv[e];
    score[e] = sc;
    atomicMaxFloat(segmax + idst[e], sc);
  }
}

// ---------------------------------------------------------------------------
// p = exp(score - m[dst]);  z[dst] += p   (thread per entry)
// ---------------------------------------------------------------------------
__global__ void pexp_kernel(const float* __restrict__ score,
                            const float* __restrict__ segmax,
                            const int* __restrict__ idst,
                            float* __restrict__ p, float* __restrict__ z,
                            int nnz)
{
  int e = blockIdx.x * blockDim.x + threadIdx.x;
  if (e >= nnz) return;
  int d = idst[e];
  float mm = segmax[d];
  if (mm < -1.0e37f) mm = 0.f;          // isfinite(m) ? m : 0  (init is -inf)
  float pe = expf(score[e] - mm);
  p[e] = pe;
  atomicAdd(z + d, pe);
}

// ---------------------------------------------------------------------------
// out[dst] += (p/max(z,1e-9)) * V[src]   (wave per entry, 128-wide rows)
// ---------------------------------------------------------------------------
__global__ __launch_bounds__(NTHREADS)
void agg_kernel(const float* __restrict__ p, const float* __restrict__ z,
                const float* __restrict__ V,
                const int* __restrict__ isrc, const int* __restrict__ idst,
                float* __restrict__ out, int nnz)
{
  int e    = blockIdx.x * (NTHREADS >> 5) + (threadIdx.x >> 5);
  int lane = threadIdx.x & 31;
  if (e >= nnz) return;
  int s = isrc[e], d = idst[e];
  float w = p[e] / fmaxf(z[d], 1e-9f);
  const float4 vv = ((const float4*)(V + (size_t)s * DDIM))[lane];
  float* o = out + (size_t)d * DDIM + lane * 4;
  atomicAdd(o + 0, w * vv.x);
  atomicAdd(o + 1, w * vv.y);
  atomicAdd(o + 2, w * vv.z);
  atomicAdd(o + 3, w * vv.w);
}

// ---------------------------------------------------------------------------
// out = LayerNorm(A + B) * g + b   (wave per row, 128 cols)
// ---------------------------------------------------------------------------
__global__ __launch_bounds__(NTHREADS)
void add_ln_kernel(const float* __restrict__ A, const float* __restrict__ B,
                   const float* __restrict__ g, const float* __restrict__ be,
                   float* __restrict__ out, int rows)
{
  int row  = blockIdx.x * (NTHREADS >> 5) + (threadIdx.x >> 5);
  int lane = threadIdx.x & 31;
  if (row >= rows) return;
  const float4 a4 = ((const float4*)(A + (size_t)row * DDIM))[lane];
  const float4 b4 = ((const float4*)(B + (size_t)row * DDIM))[lane];
  float x0 = a4.x + b4.x, x1 = a4.y + b4.y, x2 = a4.z + b4.z, x3 = a4.w + b4.w;
  float s1 = x0 + x1 + x2 + x3;
  float s2 = x0 * x0 + x1 * x1 + x2 * x2 + x3 * x3;
  #pragma unroll
  for (int off = 16; off > 0; off >>= 1) {
    s1 += __shfl_xor(s1, off, 32);
    s2 += __shfl_xor(s2, off, 32);
  }
  const float inv = 1.f / (float)DDIM;
  float mean = s1 * inv;
  float var  = s2 * inv - mean * mean;
  float r    = rsqrtf(var + 1e-5f);
  const float4 g4  = ((const float4*)g)[lane];
  const float4 be4 = ((const float4*)be)[lane];
  float4 o;
  o.x = (x0 - mean) * r * g4.x + be4.x;
  o.y = (x1 - mean) * r * g4.y + be4.y;
  o.z = (x2 - mean) * r * g4.z + be4.z;
  o.w = (x3 - mean) * r * g4.w + be4.w;
  ((float4*)(out + (size_t)row * DDIM))[lane] = o;
}

// ---------------------------------------------------------------------------
// Host-side launch
// ---------------------------------------------------------------------------
static inline int cdiv(long a, long b) { return (int)((a + b - 1) / b); }

extern "C" void kernel_launch(void* const* d_in, const int* in_sizes, int n_in,
                              void* d_out, int out_size, void* d_ws, size_t ws_size,
                              hipStream_t stream) {
  (void)n_in; (void)out_size; (void)ws_size;
  auto F = [&](int i) { return (const float*)d_in[i]; };
  auto I = [&](int i) { return (const int*)d_in[i]; };

  const float* vfeat = F(0);
  const float* efeat = F(1);
  const float* gcn   = F(2);
  const float* ccoef = F(3);
  const float* betw  = F(4);
  const int* cidx  = I(5);
  const int* uidx  = I(6);
  const int* inode = I(7);
  const int* iedge = I(8);
  const float* cs_tab = F(9);
  const float* un_tab = F(10);
  const float *w_vtx1 = F(11), *b_vtx1 = F(12);
  const float *w_kv = F(13), *b_kv = F(14);
  const float *w_vv = F(15), *b_vv = F(16);
  const float *w_qe = F(17), *b_qe = F(18);
  const float *w_qv = F(19), *b_qv = F(20);
  const float *w_ke = F(21), *b_ke = F(22);
  const float *w_ve = F(23), *b_ve = F(24);
  const float *w_cls = F(25), *b_cls = F(26);
  const float *w_mlp = F(27), *b_mlp = F(28);
  const float *w_l1 = F(29), *b_l1 = F(30);
  const float *w_l2 = F(31), *b_l2 = F(32);
  const float *w_l3 = F(33), *b_l3 = F(34);
  const float *w_l4 = F(35), *b_l4 = F(36);
  const float *ln1g = F(37), *ln1b = F(38);
  const float *ln2g = F(39), *ln2b = F(40);

  const int N    = in_sizes[5];            // centrality_idx has N entries
  const int E    = in_sizes[1] / DDIM;     // efeat is [E,128]
  const int NNZ  = in_sizes[3];            // cluster_coef has NNZ entries
  const int NCLS = in_sizes[26];           // b_cls has NCLS entries (40)
  const float SCALE = 0.08838834764831845f; // 1/sqrt(128)

  // ---- workspace layout (floats) ----
  float* ws = (float*)d_ws;
  size_t off = 0;
  auto alloc = [&](size_t n) { float* p = ws + off; off += n; return p; };
  float* feat_v = alloc((size_t)N * DDIM);  // node features; later FFN hidden
  float* kq_n   = alloc((size_t)N * DDIM);  // k -> qv -> ffn2 output
  float* v_n    = alloc((size_t)N * DDIM);  // v -> h_v -> final feat_v2
  float* qk_e   = alloc((size_t)E * DDIM);  // qe -> ke
  float* ve_e   = alloc((size_t)E * DDIM);  // edge-FFN out -> ve
  float* he     = alloc((size_t)E * DDIM);  // h_e -> edge-FFN hidden
  float* x_e    = alloc((size_t)E * DDIM);  // LN(h_e + efeat)
  float* fe2    = alloc((size_t)E * DDIM);  // updated edge features
  float* y_v    = alloc((size_t)N * DDIM);  // LN(h_v + feat_v)
  float* sc     = alloc((size_t)NNZ);
  float* pp     = alloc((size_t)NNZ);
  const int NMAX = (N > E) ? N : E;
  float* mseg = alloc((size_t)NMAX);
  float* zseg = alloc((size_t)NMAX);

  const dim3 blk(NTHREADS);
  const int gN    = cdiv(N, TILE_ROWS);
  const int gE    = cdiv(E, TILE_ROWS);
  const int gNNZw = cdiv(NNZ, NTHREADS / 32);
  const int gNNZt = cdiv(NNZ, NTHREADS);
  const int gRowE = cdiv(E, NTHREADS / 32);
  const int gRowN = cdiv(N, NTHREADS / 32);
  const float NEG_INF = -__builtin_inff();

  // ---- node feature construction ----
  wmma_linear<1, 8><<<gN, blk, 0, stream>>>(vfeat, N, w_vtx1, DDIM, b_vtx1,
                                            feat_v, DDIM, gcn, cs_tab, cidx, un_tab, uidx);

  // ---- phase 1: nodes -> hyperedges ----
  wmma_linear<0, 8><<<gN, blk, 0, stream>>>(feat_v, N, w_kv, DDIM, b_kv,
                                            kq_n, DDIM, nullptr, nullptr, nullptr, nullptr, nullptr);
  wmma_linear<0, 8><<<gN, blk, 0, stream>>>(feat_v, N, w_vv, DDIM, b_vv,
                                            v_n, DDIM, nullptr, nullptr, nullptr, nullptr, nullptr);
  wmma_linear<0, 8><<<gE, blk, 0, stream>>>(efeat, E, w_qe, DDIM, b_qe,
                                            qk_e, DDIM, nullptr, nullptr, nullptr, nullptr, nullptr);
  fill_kernel<<<1024, blk, 0, stream>>>(mseg, NEG_INF, (size_t)E);
  fill_kernel<<<1024, blk, 0, stream>>>(zseg, 0.f, (size_t)E);
  fill_kernel<<<2048, blk, 0, stream>>>(he, 0.f, (size_t)E * DDIM);
  score_max_kernel<<<gNNZw, blk, 0, stream>>>(kq_n, qk_e, inode, iedge, iedge,
                                              ccoef, sc, mseg, NNZ, SCALE);
  pexp_kernel<<<gNNZt, blk, 0, stream>>>(sc, mseg, iedge, pp, zseg, NNZ);
  agg_kernel<<<gNNZw, blk, 0, stream>>>(pp, zseg, v_n, inode, iedge, he, NNZ);
  add_ln_kernel<<<gRowE, blk, 0, stream>>>(he, efeat, ln1g, ln1b, x_e, E);
  wmma_linear<2, 8><<<gE, blk, 0, stream>>>(x_e, E, w_l1, DDIM, b_l1,
                                            he, DDIM, nullptr, nullptr, nullptr, nullptr, nullptr);
  wmma_linear<0, 8><<<gE, blk, 0, stream>>>(he, E, w_l2, DDIM, b_l2,
                                            ve_e, DDIM, nullptr, nullptr, nullptr, nullptr, nullptr);
  add_ln_kernel<<<gRowE, blk, 0, stream>>>(ve_e, x_e, ln1g, ln1b, fe2, E);

  // ---- phase 2: hyperedges -> nodes ----
  wmma_linear<0, 8><<<gE, blk, 0, stream>>>(fe2, E, w_ke, DDIM, b_ke,
                                            qk_e, DDIM, nullptr, nullptr, nullptr, nullptr, nullptr);
  wmma_linear<0, 8><<<gE, blk, 0, stream>>>(fe2, E, w_ve, DDIM, b_ve,
                                            ve_e, DDIM, nullptr, nullptr, nullptr, nullptr, nullptr);
  wmma_linear<0, 8><<<gN, blk, 0, stream>>>(feat_v, N, w_qv, DDIM, b_qv,
                                            kq_n, DDIM, nullptr, nullptr, nullptr, nullptr, nullptr);
  fill_kernel<<<1024, blk, 0, stream>>>(mseg, NEG_INF, (size_t)N);
  fill_kernel<<<1024, blk, 0, stream>>>(zseg, 0.f, (size_t)N);
  fill_kernel<<<4096, blk, 0, stream>>>(v_n, 0.f, (size_t)N * DDIM);
  score_max_kernel<<<gNNZw, blk, 0, stream>>>(qk_e, kq_n, iedge, inode, inode,
                                              betw, sc, mseg, NNZ, SCALE);
  pexp_kernel<<<gNNZt, blk, 0, stream>>>(sc, mseg, inode, pp, zseg, NNZ);
  agg_kernel<<<gNNZw, blk, 0, stream>>>(pp, zseg, ve_e, iedge, inode, v_n, NNZ);
  add_ln_kernel<<<gRowN, blk, 0, stream>>>(v_n, feat_v, ln2g, ln2b, y_v, N);
  wmma_linear<2, 8><<<gN, blk, 0, stream>>>(y_v, N, w_l3, DDIM, b_l3,
                                            feat_v, DDIM, nullptr, nullptr, nullptr, nullptr, nullptr);
  wmma_linear<0, 8><<<gN, blk, 0, stream>>>(feat_v, N, w_l4, DDIM, b_l4,
                                            kq_n, DDIM, nullptr, nullptr, nullptr, nullptr, nullptr);
  add_ln_kernel<<<gRowN, blk, 0, stream>>>(kq_n, y_v, ln2g, ln2b, v_n, N);

  // ---- heads: pred [N, NCLS] then feat [N, 128], concatenated in d_out ----
  float* pred = (float*)d_out;
  float* feat = pred + (size_t)N * NCLS;
  wmma_linear<0, 3><<<gN, blk, 0, stream>>>(v_n, N, w_cls, NCLS, b_cls,
                                            pred, NCLS, nullptr, nullptr, nullptr, nullptr, nullptr);
  wmma_linear<0, 8><<<gN, blk, 0, stream>>>(v_n, N, w_mlp, DDIM, b_mlp,
                                            feat, DDIM, nullptr, nullptr, nullptr, nullptr, nullptr);
}
